// MoEModel_7593502179956
// MI455X (gfx1250) — matrix operand order repaired and checked by reading, not compiled, not run on hardware
//
#include <hip/hip_runtime.h>
#include <hip/hip_bf16.h>
#include <math.h>

typedef __bf16 bhalf_t;
typedef __attribute__((ext_vector_type(16))) __bf16 v16bf;
typedef __attribute__((ext_vector_type(8)))  __bf16 v8bf;
typedef __attribute__((ext_vector_type(2)))  __bf16 v2bf;
typedef __attribute__((ext_vector_type(8)))  float  v8f;

#define DEVINL __device__ __forceinline__

DEVINL float gelu_exact(float x){
    return 0.5f * x * (1.0f + erff(x * 0.70710678118654752440f));
}

DEVINL v8f wmma_bf16(v16bf a, v16bf b, v8f c){
    // D = A(16x32 bf16) x B(32x16 bf16) + C(16x16 f32)
    return __builtin_amdgcn_wmma_f32_16x16x32_bf16(false, a, false, b, (short)0, c, false, false);
}

// A-fragment (16x32 bf16 tile) from LDS with row stride SAS elements.
template<int SAS>
DEVINL v16bf load_a_frag(const bhalf_t* sa, int lane){
    const int row = lane & 15;
    const int kb  = (lane >> 4) << 3;
    const bhalf_t* p = sa + row * SAS + kb;
    v8bf lo = *reinterpret_cast<const v8bf*>(p);
    v8bf hi = *reinterpret_cast<const v8bf*>(p + 16);
    v16bf a;
    #pragma unroll
    for (int e = 0; e < 8; ++e){ a[e] = lo[e]; a[8 + e] = hi[e]; }
    return a;
}

// B-fragment (32x16 tile at column `col`) from a TRANSPOSED LDS slab
// swt[col][k] (32 contiguous k per column) -> two contiguous ds_load_b128.
DEVINL v16bf load_b_frag_t(const bhalf_t* swt, int col, int lane){
    const int kb = (lane >> 4) << 3;
    const bhalf_t* p = swt + col * 32 + kb;
    v8bf lo = *reinterpret_cast<const v8bf*>(p);
    v8bf hi = *reinterpret_cast<const v8bf*>(p + 16);
    v16bf b;
    #pragma unroll
    for (int e = 0; e < 8; ++e){ b[e] = lo[e]; b[8 + e] = hi[e]; }
    return b;
}

// Column-wise W staging: thread owns column c, walks 32 k-values (row reads stay
// lane-coalesced), packs bf16, writes the column as 4x ds_store_b128.
// One exec guard per 256-column group (elided when the group is full).
template<int N>
DEVINL void stage_w_cols(const float* __restrict__ W, bhalf_t* __restrict__ dstSlab,
                         int k0, int tid){
    #pragma unroll
    for (int c0 = 0; c0 < N; c0 += 256){
        const int c = c0 + tid;
        if ((N - c0 >= 256) || (c < N)){
            const float* src = W + (size_t)k0 * N + c;
            v8bf colv[4];
            #pragma unroll
            for (int q = 0; q < 4; ++q)
                #pragma unroll
                for (int e = 0; e < 8; ++e)
                    colv[q][e] = (bhalf_t)src[(size_t)(q * 8 + e) * N];
            v8bf* dst = reinterpret_cast<v8bf*>(dstSlab + c * 32);
            #pragma unroll
            for (int q = 0; q < 4; ++q) dst[q] = colv[q];
        }
    }
}

struct Ptr5 { const float* p[5]; };

// ---------------------------------------------------------------------------
// Fused projection: Y[b,n,:] = LN(GELU(X_n[b,:] @ W_n + bias_n)), H=384, D=768
// 256 threads (8 waves); block = 16 rows x 384 cols; ping-pong LDS staging.
// ---------------------------------------------------------------------------
__global__ __launch_bounds__(256)
void proj_kernel(Ptr5 xs,
                 const float* __restrict__ Wb,   // Nmat x 768 x 384
                 const float* __restrict__ bb,   // Nmat x 384
                 const float* __restrict__ gb,
                 const float* __restrict__ beb,
                 float*   __restrict__ outF,     // (B, Nmat, 384) or null
                 bhalf_t* __restrict__ outB,     // (B, Nmat, 384)
                 int Nmat)
{
    const int D = 768, H = 384, NSTEP = D / 32;
    const int n    = blockIdx.y;
    const int row0 = blockIdx.x * 16;
    const float* __restrict__ X  = xs.p[n];
    const float* __restrict__ W  = Wb  + (size_t)n * D * H;
    const float* __restrict__ bi = bb  + (size_t)n * H;
    const float* __restrict__ lg = gb  + (size_t)n * H;
    const float* __restrict__ lb = beb + (size_t)n * H;

    __shared__ bhalf_t sa[2][16 * 32];
    __shared__ bhalf_t swt[2][384 * 32];   // transposed: [col][k]
    __shared__ float   sy[16 * 384];
    __shared__ float   red0[256], red1[256];
    __shared__ float   smean[16], srstd[16];

    const int tid = threadIdx.x, lane = tid & 31, wave = tid >> 5;

    auto stage = [&](int buf, int k0){
        // A tile: thread loads a float2, packed-converts, 4-byte LDS store.
        {
            int r = tid >> 4, c = (tid & 15) << 1;
            float2 xv = *reinterpret_cast<const float2*>(X + (size_t)(row0 + r) * D + k0 + c);
            v2bf pk; pk[0] = (bhalf_t)xv.x; pk[1] = (bhalf_t)xv.y;
            *reinterpret_cast<v2bf*>(&sa[buf][r * 32 + c]) = pk;
        }
        stage_w_cols<384>(W, swt[buf], k0, tid);
        if (tid < 32 && k0 + 64 < D)
            __builtin_prefetch(W + (size_t)(k0 + 64 + tid) * H, 0, 1);
    };

    v8f acc[3];
    #pragma unroll
    for (int i = 0; i < 3; ++i)
        #pragma unroll
        for (int e = 0; e < 8; ++e) acc[i][e] = 0.f;

    stage(0, 0);
    __syncthreads();
    #pragma unroll 2
    for (int it = 0; it < NSTEP; ++it){
        const int cur = it & 1;
        if (it + 1 < NSTEP) stage(cur ^ 1, (it + 1) * 32);
        v16bf af = load_a_frag<32>(sa[cur], lane);
        #pragma unroll
        for (int i = 0; i < 3; ++i){
            int col = (wave * 3 + i) * 16 + (lane & 15);
            v16bf bfr = load_b_frag_t(swt[cur], col, lane);
            acc[i] = wmma_bf16(af, bfr, acc[i]);
        }
        __syncthreads();
    }

    // bias + GELU into LDS (C layout: vgpr e, lane -> row e+8*(lane/16), col lane&15)
    #pragma unroll
    for (int i = 0; i < 3; ++i){
        int col = (wave * 3 + i) * 16 + (lane & 15);
        #pragma unroll
        for (int e = 0; e < 8; ++e){
            int r = e + ((lane >> 4) << 3);
            sy[r * 384 + col] = gelu_exact(acc[i][e] + bi[col]);
        }
    }
    __syncthreads();

    // LayerNorm: 16 rows x 16 threads each
    const int r = tid >> 4, tx = tid & 15;
    float s = 0.f, s2 = 0.f;
    #pragma unroll
    for (int c0 = 0; c0 < H; c0 += 16){ float v = sy[r * 384 + c0 + tx]; s += v; s2 += v * v; }
    red0[tid] = s; red1[tid] = s2;
    __syncthreads();
    if (tx == 0){
        float a = 0.f, b = 0.f;
        #pragma unroll
        for (int j = 0; j < 16; ++j){ a += red0[(r << 4) + j]; b += red1[(r << 4) + j]; }
        float m   = a * (1.0f / 384.0f);
        float var = b * (1.0f / 384.0f) - m * m;
        smean[r] = m; srstd[r] = rsqrtf(var + 1e-5f);
    }
    __syncthreads();
    {
        float m = smean[r], rr = srstd[r];
        size_t orow = ((size_t)(row0 + r) * Nmat + n) * H;
        for (int c = tx; c < H; c += 16){
            float v = (sy[r * 384 + c] - m) * rr * lg[c] + lb[c];
            if (outF) outF[orow + c] = v;
            outB[orow + c] = (bhalf_t)v;
        }
    }
}

// ---------------------------------------------------------------------------
// Routing: rlog[b,r,n] = sum_m GELU((bs @ route_w1[r]) + b1)[.,m] * w2[r,m] + b2[r]
// bs rows are (b*4+n). Block: 16 b's (64 rows) x one r; t never materialized.
// ---------------------------------------------------------------------------
__global__ __launch_bounds__(256)
void routing_kernel(const bhalf_t* __restrict__ bs, // (B*4, 384)
                    const float* __restrict__ w1,   // (4, 384, 192)
                    const float* __restrict__ b1,   // (4, 192)
                    const float* __restrict__ w2,   // (4, 192)
                    const float* __restrict__ b2,   // (4)
                    float* __restrict__ rlog)       // (B, 4, 4)
{
    const int H = 384, M = 192, NSTEP = H / 32;
    const int r    = blockIdx.y;
    const int row0 = blockIdx.x * 64;

    __shared__ bhalf_t sa[2][64 * 32];
    __shared__ bhalf_t swt[2][192 * 32];   // transposed
    __shared__ float   sw2[192];
    __shared__ float   srl[64];

    const int tid = threadIdx.x, lane = tid & 31, wave = tid >> 5;
    if (tid < 192) sw2[tid] = w2[r * 192 + tid];
    if (tid < 64)  srl[tid] = 0.f;

    const float* __restrict__ W = w1 + (size_t)r * H * M;
    const int rt  = wave >> 1;        // row tile 0..3
    const int cb0 = (wave & 1) * 6;   // first col tile (6 per wave)

    auto stage = [&](int buf, int k0){
        {   // A: 64x32 bf16 = exactly 256 16-byte chunks
            int rr = tid >> 2, cc = (tid & 3) << 3;
            *reinterpret_cast<v8bf*>(&sa[buf][rr * 32 + cc]) =
                *reinterpret_cast<const v8bf*>(bs + (size_t)(row0 + rr) * H + k0 + cc);
        }
        stage_w_cols<192>(W, swt[buf], k0, tid);
    };

    v8f acc[6];
    #pragma unroll
    for (int i = 0; i < 6; ++i)
        #pragma unroll
        for (int e = 0; e < 8; ++e) acc[i][e] = 0.f;

    stage(0, 0);
    __syncthreads();
    #pragma unroll 2
    for (int it = 0; it < NSTEP; ++it){
        const int cur = it & 1;
        if (it + 1 < NSTEP) stage(cur ^ 1, (it + 1) * 32);
        v16bf af = load_a_frag<32>(&sa[cur][rt * 16 * 32], lane);
        #pragma unroll
        for (int i = 0; i < 6; ++i){
            int col = (cb0 + i) * 16 + (lane & 15);
            v16bf bfr = load_b_frag_t(swt[cur], col, lane);
            acc[i] = wmma_bf16(af, bfr, acc[i]);
        }
        __syncthreads();
    }

    // fused GELU + dot with w2, reduced over the 192 columns
    float rs[8];
    #pragma unroll
    for (int e = 0; e < 8; ++e) rs[e] = 0.f;
    #pragma unroll
    for (int i = 0; i < 6; ++i){
        int col = (cb0 + i) * 16 + (lane & 15);
        float bias = b1[r * 192 + col];
        float wv   = sw2[col];
        #pragma unroll
        for (int e = 0; e < 8; ++e)
            rs[e] += gelu_exact(acc[i][e] + bias) * wv;
    }
    #pragma unroll
    for (int off = 1; off < 16; off <<= 1)
        #pragma unroll
        for (int e = 0; e < 8; ++e) rs[e] += __shfl_xor(rs[e], off, 32);
    if ((lane & 15) == 0){
        #pragma unroll
        for (int e = 0; e < 8; ++e){
            int row = rt * 16 + e + ((lane >> 4) << 3);
            atomicAdd(&srl[row], rs[e]);
        }
    }
    __syncthreads();
    if (tid < 64){
        int grow = row0 + tid;
        int b = grow >> 2, nn = grow & 3;
        rlog[(size_t)b * 16 + r * 4 + nn] = srl[tid] + b2[r];
    }
}

// ---------------------------------------------------------------------------
// Softmax over n, heads = att . base_stack, ctx = mean_n.  One b per block.
// ---------------------------------------------------------------------------
__global__ void heads_kernel(const float* __restrict__ rlog, // (B,16)
                             const float* __restrict__ bsf,  // (B*4,384)
                             float* __restrict__ fused,      // (B,2688)
                             bhalf_t* __restrict__ ctxb)     // (B,384)
{
    const int b = blockIdx.x;
    const float* rl = rlog + (size_t)b * 16;
    float att[4][4];
    #pragma unroll
    for (int r = 0; r < 4; ++r){
        float m = rl[r * 4];
        #pragma unroll
        for (int n = 1; n < 4; ++n) m = fmaxf(m, rl[r * 4 + n]);
        float s = 0.f;
        #pragma unroll
        for (int n = 0; n < 4; ++n){ att[r][n] = __expf(rl[r * 4 + n] - m); s += att[r][n]; }
        float inv = 1.f / s;
        #pragma unroll
        for (int n = 0; n < 4; ++n) att[r][n] *= inv;
    }
    float* fr = fused + (size_t)b * 2688;
    const float* x = bsf + (size_t)b * 4 * 384;
    for (int h = threadIdx.x; h < 384; h += blockDim.x){
        float v0 = x[h], v1 = x[384 + h], v2 = x[768 + h], v3 = x[1152 + h];
        float ctx = 0.25f * (v0 + v1 + v2 + v3);
        #pragma unroll
        for (int r = 0; r < 4; ++r)
            fr[r * 384 + h] = att[r][0] * v0 + att[r][1] * v1 + att[r][2] * v2 + att[r][3] * v3;
        fr[1536 + h] = ctx;
        ctxb[(size_t)b * 384 + h] = (bhalf_t)ctx;
    }
}

// Build concat inputs for warm ([we,tg,ctx]) and comp ([ce,tg,ctx]) in bf16.
__global__ void pack_kernel(const bhalf_t* __restrict__ fh,  // (B,5,384)
                            const bhalf_t* __restrict__ ctx, // (B,384)
                            bhalf_t* __restrict__ Xw,        // (B,1152)
                            bhalf_t* __restrict__ Xc)        // (B,1152)
{
    const int b = blockIdx.x;
    const bhalf_t* f = fh  + (size_t)b * 5 * 384;
    const bhalf_t* c = ctx + (size_t)b * 384;
    bhalf_t* w = Xw + (size_t)b * 1152;
    bhalf_t* q = Xc + (size_t)b * 1152;
    for (int i = threadIdx.x; i < 384; i += blockDim.x){
        bhalf_t tg = f[i], we = f[384 + i], ce = f[768 + i], ct = c[i];
        w[i] = we; w[384 + i] = tg; w[768 + i] = ct;
        q[i] = ce; q[384 + i] = tg; q[768 + i] = ct;
    }
}

// ---------------------------------------------------------------------------
// Generic WMMA GEMM: out = [LN]( [GELU]( A_bf16 @ W_f32 + bias ) )
// All dims compile-time. 256 threads; 16 rows x N cols; wave owns NT tiles.
// ---------------------------------------------------------------------------
template<int NT, bool ACT_GELU, bool DO_LN, int N, int K, int LDA>
__global__ __launch_bounds__(256)
void gemm_kernel(const bhalf_t* __restrict__ A,
                 const float* __restrict__ W,      // K x N row-major
                 const float* __restrict__ bias,   // N
                 const float* __restrict__ ln_g,   // N (LN only)
                 const float* __restrict__ ln_b,   // N (LN only)
                 float*   __restrict__ outF, int ldof,
                 bhalf_t* __restrict__ outB, int ldob)
{
    constexpr int NSTEP = K / 32;
    __shared__ bhalf_t sa[2][16 * 32];
    __shared__ bhalf_t swt[2][N * 32];  // transposed: [col][k]
    __shared__ float   sy[DO_LN ? 16 * N : 16];
    __shared__ float   red0[256], red1[256];
    __shared__ float   smean[16], srstd[16];

    const int tid = threadIdx.x, lane = tid & 31, wave = tid >> 5;
    const int row0 = blockIdx.x * 16;
    constexpr int ntiles = (N + 15) >> 4;

    v8f acc[NT];
    #pragma unroll
    for (int i = 0; i < NT; ++i)
        #pragma unroll
        for (int e = 0; e < 8; ++e) acc[i][e] = 0.f;

    const bhalf_t* Arow = A + (size_t)row0 * LDA;

    auto stage = [&](int buf, int k0){
        if (tid < 64){   // A: 16x32 bf16 = 64 16-byte chunks
            int r = tid >> 2, cc = (tid & 3) << 3;
            *reinterpret_cast<v8bf*>(&sa[buf][r * 32 + cc]) =
                *reinterpret_cast<const v8bf*>(Arow + (size_t)r * LDA + k0 + cc);
        }
        stage_w_cols<N>(W, swt[buf], k0, tid);
        if (tid < 32 && k0 + 64 < K)
            __builtin_prefetch(W + (size_t)(k0 + 64 + tid) * N, 0, 1);
    };

    stage(0, 0);
    __syncthreads();
    #pragma unroll 2
    for (int it = 0; it < NSTEP; ++it){
        const int cur = it & 1;
        if (it + 1 < NSTEP) stage(cur ^ 1, (it + 1) * 32);
        v16bf af = load_a_frag<32>(sa[cur], lane);
        #pragma unroll
        for (int i = 0; i < NT; ++i){
            int tile = wave * NT + i;
            if (tile < ntiles){             // wave-uniform; EXEC stays full for WMMA
                int col = tile * 16 + (lane & 15);
                int colc = col < N ? col : N - 1;
                v16bf bfr = load_b_frag_t(swt[cur], colc, lane);
                acc[i] = wmma_bf16(af, bfr, acc[i]);
            }
        }
        __syncthreads();
    }

    if (!DO_LN){
        #pragma unroll
        for (int i = 0; i < NT; ++i){
            int tile = wave * NT + i;
            if (tile < ntiles){
                int col = tile * 16 + (lane & 15);
                #pragma unroll
                for (int e = 0; e < 8; ++e){
                    int row = row0 + e + ((lane >> 4) << 3);
                    if (col < N){
                        float v = acc[i][e] + bias[col];
                        if (ACT_GELU) v = gelu_exact(v);
                        if (outF) outF[(size_t)row * ldof + col] = v;
                        if (outB) outB[(size_t)row * ldob + col] = (bhalf_t)v;
                    }
                }
            }
        }
    } else {
        #pragma unroll
        for (int i = 0; i < NT; ++i){
            int tile = wave * NT + i;
            if (tile < ntiles){
                int col = tile * 16 + (lane & 15);
                #pragma unroll
                for (int e = 0; e < 8; ++e){
                    int r = e + ((lane >> 4) << 3);
                    float v = acc[i][e] + bias[col];
                    if (ACT_GELU) v = gelu_exact(v);
                    sy[r * N + col] = v;
                }
            }
        }
        __syncthreads();
        const int r = tid >> 4, tx = tid & 15;
        float s = 0.f, s2 = 0.f;
        #pragma unroll
        for (int c0 = 0; c0 < N; c0 += 16){ float v = sy[r * N + c0 + tx]; s += v; s2 += v * v; }
        red0[tid] = s; red1[tid] = s2;
        __syncthreads();
        if (tx == 0){
            float a = 0.f, b = 0.f;
            #pragma unroll
            for (int j = 0; j < 16; ++j){ a += red0[(r << 4) + j]; b += red1[(r << 4) + j]; }
            float m = a / (float)N;
            float var = b / (float)N - m * m;
            smean[r] = m; srstd[r] = rsqrtf(var + 1e-5f);
        }
        __syncthreads();
        float m = smean[r], rr = srstd[r];
        for (int c = tx; c < N; c += 16){
            float v = (sy[r * N + c] - m) * rr * ln_g[c] + ln_b[c];
            size_t row = (size_t)(row0 + r);
            if (outF) outF[row * ldof + c] = v;
            if (outB) outB[row * ldob + c] = (bhalf_t)v;
        }
    }
}

// Router softmax + quadrant = dist @ protos. One b per block.
__global__ void router_kernel(const float* __restrict__ rlogit, // (B,8)
                              const float* __restrict__ protos, // (8,384)
                              float* __restrict__ rdist,        // (B,8)
                              float* __restrict__ fused)        // (B,2688), cols 1920..2303
{
    const int b = blockIdx.x;
    const float* L = rlogit + (size_t)b * 8;
    float d[8];
    float m = L[0];
    #pragma unroll
    for (int e = 1; e < 8; ++e) m = fmaxf(m, L[e]);
    float s = 0.f;
    #pragma unroll
    for (int e = 0; e < 8; ++e){ d[e] = __expf(L[e] - m); s += d[e]; }
    float inv = 1.f / s;
    #pragma unroll
    for (int e = 0; e < 8; ++e) d[e] *= inv;
    if (threadIdx.x < 8) rdist[(size_t)b * 8 + threadIdx.x] = d[threadIdx.x];
    float* fq = fused + (size_t)b * 2688 + 1920;
    for (int h = threadIdx.x; h < 384; h += blockDim.x){
        float acc = 0.f;
        #pragma unroll
        for (int e = 0; e < 8; ++e) acc += d[e] * protos[e * 384 + h];
        fq[h] = acc;
    }
}

// Row LayerNorm over C=2688 columns -> bf16. One b per block of 256.
__global__ __launch_bounds__(256)
void ln_row_kernel(const float* __restrict__ x, const float* __restrict__ g,
                   const float* __restrict__ be, bhalf_t* __restrict__ out, int C)
{
    const int b = blockIdx.x, tid = threadIdx.x;
    const float* xr = x + (size_t)b * C;
    __shared__ float r1[256], r2[256];
    float s = 0.f, s2 = 0.f;
    for (int c = tid; c < C; c += 256){ float v = xr[c]; s += v; s2 += v * v; }
    r1[tid] = s; r2[tid] = s2;
    __syncthreads();
    for (int o = 128; o > 0; o >>= 1){
        if (tid < o){ r1[tid] += r1[tid + o]; r2[tid] += r2[tid + o]; }
        __syncthreads();
    }
    float m = r1[0] / (float)C;
    float var = r2[0] / (float)C - m * m;
    float rr = rsqrtf(var + 1e-5f);
    bhalf_t* po = out + (size_t)b * C;
    for (int c = tid; c < C; c += 256) po[c] = (bhalf_t)((xr[c] - m) * rr * g[c] + be[c]);
}

// ---------------------------------------------------------------------------
// Experts: one wave per expert per 16-row tile. eh = GELU(shared @ ex_w1[x]+b1),
// expert_out = eh @ ex_w2[x] + b2, logits = sum_x dist[b,x] * expert_out.
// ---------------------------------------------------------------------------
__global__ __launch_bounds__(256)
void expert_kernel(const bhalf_t* __restrict__ sh,  // (B,128)
                   const float* __restrict__ w1,    // (8,128,64)
                   const float* __restrict__ b1,    // (8,64)
                   const float* __restrict__ w2,    // (8,64,2)
                   const float* __restrict__ b2,    // (8,2)
                   const float* __restrict__ rdist, // (B,8)
                   float* __restrict__ out)         // (B,2)
{
    const int row0 = blockIdx.x * 16;
    __shared__ bhalf_t sa[16 * 128];
    __shared__ bhalf_t swt[8][64 * 32];  // per-expert transposed W slab (wave-private)
    __shared__ float   so[32];
    const int tid = threadIdx.x, lane = tid & 31, xp = tid >> 5; // expert id

    // stage A (16x128 bf16): 256 16-byte chunks
    {
        int r = tid >> 4, cc = (tid & 15) << 3;
        *reinterpret_cast<v8bf*>(&sa[r * 128 + cc]) =
            *reinterpret_cast<const v8bf*>(sh + (size_t)(row0 + r) * 128 + cc);
    }
    if (tid < 32) so[tid] = 0.f;
    __syncthreads();

    const float* __restrict__ W = w1 + (size_t)xp * 128 * 64;
    bhalf_t* myw = &swt[xp][0];
    v8f acc[4];
    #pragma unroll
    for (int i = 0; i < 4; ++i)
        #pragma unroll
        for (int e = 0; e < 8; ++e) acc[i][e] = 0.f;

    #pragma unroll 1
    for (int k0 = 0; k0 < 128; k0 += 32){
        // wave-private column-wise transposed staging of this expert's 32x64 W slab
        #pragma unroll
        for (int c0 = 0; c0 < 64; c0 += 32){
            const int c = c0 + lane;
            const float* src = W + (size_t)k0 * 64 + c;
            v8bf colv[4];
            #pragma unroll
            for (int q = 0; q < 4; ++q)
                #pragma unroll
                for (int e = 0; e < 8; ++e)
                    colv[q][e] = (bhalf_t)src[(size_t)(q * 8 + e) * 64];
            v8bf* dst = reinterpret_cast<v8bf*>(myw + c * 32);
            #pragma unroll
            for (int q = 0; q < 4; ++q) dst[q] = colv[q];
        }
        v16bf af = load_a_frag<128>(sa + k0, lane);
        #pragma unroll
        for (int i = 0; i < 4; ++i){
            int col = i * 16 + (lane & 15);
            v16bf bfr = load_b_frag_t(myw, col, lane);
            acc[i] = wmma_bf16(af, bfr, acc[i]);
        }
    }

    float rs0[8], rs1[8];
    #pragma unroll
    for (int e = 0; e < 8; ++e){ rs0[e] = 0.f; rs1[e] = 0.f; }
    #pragma unroll
    for (int i = 0; i < 4; ++i){
        int f = i * 16 + (lane & 15);
        float w20 = w2[xp * 128 + f * 2 + 0];
        float w21 = w2[xp * 128 + f * 2 + 1];
        float bb  = b1[xp * 64 + f];
        #pragma unroll
        for (int e = 0; e < 8; ++e){
            float eh = gelu_exact(acc[i][e] + bb);
            rs0[e] += eh * w20;
            rs1[e] += eh * w21;
        }
    }
    #pragma unroll
    for (int off = 1; off < 16; off <<= 1)
        #pragma unroll
        for (int e = 0; e < 8; ++e){
            rs0[e] += __shfl_xor(rs0[e], off, 32);
            rs1[e] += __shfl_xor(rs1[e], off, 32);
        }
    if ((lane & 15) == 0){
        #pragma unroll
        for (int e = 0; e < 8; ++e){
            int rr = e + ((lane >> 4) << 3);
            float wgt = rdist[(size_t)(row0 + rr) * 8 + xp];
            atomicAdd(&so[rr * 2 + 0], (rs0[e] + b2[xp * 2 + 0]) * wgt);
            atomicAdd(&so[rr * 2 + 1], (rs1[e] + b2[xp * 2 + 1]) * wgt);
        }
    }
    __syncthreads();
    if (tid < 32) out[(size_t)row0 * 2 + tid] = so[tid];
}

// ---------------------------------------------------------------------------
extern "C" void kernel_launch(void* const* d_in, const int* in_sizes, int n_in,
                              void* d_out, int out_size, void* d_ws, size_t ws_size,
                              hipStream_t stream)
{
    (void)in_sizes; (void)n_in; (void)out_size; (void)ws_size;
    const int B = 16384;

    const float* base0 = (const float*)d_in[0];
    const float* base1 = (const float*)d_in[1];
    const float* base2 = (const float*)d_in[2];
    const float* base3 = (const float*)d_in[3];
    const float* tg_in = (const float*)d_in[4];
    const float* we_in = (const float*)d_in[5];
    const float* ce_in = (const float*)d_in[6];
    const float* sp_in = (const float*)d_in[7];
    const float* bt_in = (const float*)d_in[8];

    const float* bp_w  = (const float*)d_in[9];
    const float* bp_b  = (const float*)d_in[10];
    const float* bp_g  = (const float*)d_in[11];
    const float* bp_be = (const float*)d_in[12];
    const float* fp_w  = (const float*)d_in[13];
    const float* fp_b  = (const float*)d_in[14];
    const float* fp_g  = (const float*)d_in[15];
    const float* fp_be = (const float*)d_in[16];
    const float* warm_w1 = (const float*)d_in[17];
    const float* warm_b1 = (const float*)d_in[18];
    const float* warm_w2 = (const float*)d_in[19];
    const float* warm_b2 = (const float*)d_in[20];
    const float* warm_g  = (const float*)d_in[21];
    const float* warm_be = (const float*)d_in[22];
    const float* comp_w1 = (const float*)d_in[23];
    const float* comp_b1 = (const float*)d_in[24];
    const float* comp_w2 = (const float*)d_in[25];
    const float* comp_b2 = (const float*)d_in[26];
    const float* comp_g  = (const float*)d_in[27];
    const float* comp_be = (const float*)d_in[28];
    const float* perc_w1 = (const float*)d_in[29];
    const float* perc_b1 = (const float*)d_in[30];
    const float* perc_g  = (const float*)d_in[31];
    const float* perc_be = (const float*)d_in[32];
    const float* route_w1 = (const float*)d_in[33];
    const float* route_b1 = (const float*)d_in[34];
    const float* route_w2 = (const float*)d_in[35];
    const float* route_b2 = (const float*)d_in[36];
    const float* qh_w1 = (const float*)d_in[37];
    const float* qh_b1 = (const float*)d_in[38];
    const float* qh_w2 = (const float*)d_in[39];
    const float* qh_b2 = (const float*)d_in[40];
    const float* protos = (const float*)d_in[41];
    const float* pc_g  = (const float*)d_in[42];
    const float* pc_be = (const float*)d_in[43];
    const float* pc_w1 = (const float*)d_in[44];
    const float* pc_b1 = (const float*)d_in[45];
    const float* pc_w2 = (const float*)d_in[46];
    const float* pc_b2 = (const float*)d_in[47];
    const float* ex_w1 = (const float*)d_in[48];
    const float* ex_b1 = (const float*)d_in[49];
    const float* ex_w2 = (const float*)d_in[50];
    const float* ex_b2 = (const float*)d_in[51];

    char* ws = (char*)d_ws;
    size_t off = 0;
    auto alloc = [&](size_t n) -> char* {
        char* p = ws + off;
        off += (n + 255) & ~(size_t)255;
        return p;
    };

    float*   base_f = (float*)  alloc((size_t)B * 4 * 384 * 4);  // base_stack f32
    bhalf_t* base_b = (bhalf_t*)alloc((size_t)B * 4 * 384 * 2);  // base_stack bf16
    bhalf_t* fh_b   = (bhalf_t*)alloc((size_t)B * 5 * 384 * 2);  // field stack bf16
    float*   fused  = (float*)  alloc((size_t)B * 2688 * 4);     // [heads|ctx|quad|perc]
    float*   rlog   = (float*)  alloc((size_t)B * 16 * 4);
    bhalf_t* ctx_b  = (bhalf_t*)alloc((size_t)B * 384 * 2);
    bhalf_t* Xc     = (bhalf_t*)alloc((size_t)B * 1152 * 2);
    bhalf_t* hw     = (bhalf_t*)alloc((size_t)B * 384 * 2);
    bhalf_t* hc     = (bhalf_t*)alloc((size_t)B * 384 * 2);
    bhalf_t* wc     = (bhalf_t*)alloc((size_t)B * 768 * 2);      // [warm|comp] bf16
    bhalf_t* qh_h   = (bhalf_t*)alloc((size_t)B * 384 * 2);
    float*   rlogit = (float*)  alloc((size_t)B * 8 * 4);
    float*   rdist  = (float*)  alloc((size_t)B * 8 * 4);
    bhalf_t* s1     = (bhalf_t*)alloc((size_t)B * 256 * 2);
    bhalf_t* shb    = (bhalf_t*)alloc((size_t)B * 128 * 2);
    // Buffer reuse (lifetimes disjoint on the in-order stream):
    bhalf_t* Xw       = base_b;           // base_b dead after routing_kernel
    bhalf_t* fused_ln = (bhalf_t*)base_f; // base_f dead after heads_kernel

    const dim3 blk(256);
    const dim3 g16(B / 16);

    // 1) Base + field projections (GEMM + GELU + LN), bf16 WMMA.
    Ptr5 xb; xb.p[0] = base0; xb.p[1] = base1; xb.p[2] = base2; xb.p[3] = base3; xb.p[4] = nullptr;
    proj_kernel<<<dim3(B / 16, 4), blk, 0, stream>>>(xb, bp_w, bp_b, bp_g, bp_be, base_f, base_b, 4);
    Ptr5 xf; xf.p[0] = tg_in; xf.p[1] = we_in; xf.p[2] = ce_in; xf.p[3] = sp_in; xf.p[4] = bt_in;
    proj_kernel<<<dim3(B / 16, 5), blk, 0, stream>>>(xf, fp_w, fp_b, fp_g, fp_be, nullptr, fh_b, 5);

    // 2) Routing logits (WMMA + fused w2 contraction), softmax+heads+ctx.
    routing_kernel<<<dim3(B / 16, 4), blk, 0, stream>>>(base_b, route_w1, route_b1, route_w2, route_b2, rlog);
    heads_kernel<<<dim3(B), dim3(128), 0, stream>>>(rlog, base_f, fused, ctx_b);

    // 3) warm / comp / percept / quadrant-head pipelines.
    pack_kernel<<<dim3(B), dim3(128), 0, stream>>>(fh_b, ctx_b, Xw, Xc);
    gemm_kernel<3, true,  false, 384, 1152, 1152><<<g16, blk, 0, stream>>>(Xw, warm_w1, warm_b1, nullptr, nullptr, nullptr, 0, hw, 384);
    gemm_kernel<3, true,  false, 384, 1152, 1152><<<g16, blk, 0, stream>>>(Xc, comp_w1, comp_b1, nullptr, nullptr, nullptr, 0, hc, 384);
    gemm_kernel<3, false, true,  384, 384,  384 ><<<g16, blk, 0, stream>>>(hw, warm_w2, warm_b2, warm_g, warm_be, nullptr, 0, wc,       768);
    gemm_kernel<3, false, true,  384, 384,  384 ><<<g16, blk, 0, stream>>>(hc, comp_w2, comp_b2, comp_g, comp_be, nullptr, 0, wc + 384, 768);
    gemm_kernel<3, true,  true,  384, 768,  1920><<<g16, blk, 0, stream>>>(fh_b + 3 * 384, perc_w1, perc_b1, perc_g, perc_be, fused + 2304, 2688, nullptr, 0);
    gemm_kernel<3, true,  false, 384, 768,  768 ><<<g16, blk, 0, stream>>>(wc, qh_w1, qh_b1, nullptr, nullptr, nullptr, 0, qh_h, 384);
    gemm_kernel<1, false, false, 8,   384,  384 ><<<g16, blk, 0, stream>>>(qh_h, qh_w2, qh_b2, nullptr, nullptr, rlogit, 8, nullptr, 0);
    router_kernel<<<dim3(B), dim3(128), 0, stream>>>(rlogit, protos, rdist, fused);

    // 4) pc stack: LN(fused) -> 256 -> 128 (GELU each).
    ln_row_kernel<<<dim3(B), blk, 0, stream>>>(fused, pc_g, pc_be, fused_ln, 2688);
    gemm_kernel<2, true, false, 256, 2688, 2688><<<g16, blk, 0, stream>>>(fused_ln, pc_w1, pc_b1, nullptr, nullptr, nullptr, 0, s1, 256);
    gemm_kernel<1, true, false, 128, 256,  256 ><<<g16, blk, 0, stream>>>(s1, pc_w2, pc_b2, nullptr, nullptr, nullptr, 0, shb, 128);

    // 5) Experts + router-weighted mixture -> logits (B,2).
    expert_kernel<<<g16, blk, 0, stream>>>(shb, ex_w1, ex_b1, ex_w2, ex_b2, rdist, (float*)d_out);
}
